// HGRN2Attention_89807766159672
// MI455X (gfx1250) — matrix operand order repaired and checked by reading, not compiled, not run on hardware
//
#include <hip/hip_runtime.h>
#include <hip/hip_bf16.h>
#include <math.h>

typedef __bf16 bf16x8  __attribute__((ext_vector_type(8)));
typedef __bf16 bf16x16 __attribute__((ext_vector_type(16)));
typedef float  v2f     __attribute__((ext_vector_type(2)));
typedef float  v8f     __attribute__((ext_vector_type(8)));
typedef unsigned int u32x4 __attribute__((ext_vector_type(4)));
typedef int    i32x4   __attribute__((ext_vector_type(4)));
typedef int    i32x8   __attribute__((ext_vector_type(8)));

#define D_MODEL 1024
#define SEQ_T   4096
#define NB      4
#define NH      8
#define DF      128
#define CHK     64
#define NC      (SEQ_T / CHK)
#define KT      128   // GEMM K-tile staged in LDS per TDM transfer

// ---------------------------------------------------------------------------
// fp32 -> bf16 (round to nearest even)
// ---------------------------------------------------------------------------
__device__ __forceinline__ unsigned short f2bf(float f) {
  union { float f; unsigned int u; } v; v.f = f;
  unsigned int u = v.u;
  if ((u & 0x7F800000u) == 0x7F800000u) return (unsigned short)(u >> 16); // inf/nan
  return (unsigned short)((u + 0x7FFFu + ((u >> 16) & 1u)) >> 16);
}

__global__ __launch_bounds__(256)
void cvt_bf16_kernel(const float* __restrict__ src, unsigned short* __restrict__ dst, int n) {
  for (int i = blockIdx.x * blockDim.x + threadIdx.x; i < n; i += gridDim.x * blockDim.x)
    dst[i] = f2bf(src[i]);
}

// ---------------------------------------------------------------------------
// Activations
// ---------------------------------------------------------------------------
template <int MODE>
__device__ __forceinline__ float act_fn(float v) {
  if (MODE == 0) return v / (1.f + __expf(-v));                           // silu
  if (MODE == 1) return fminf(v, 0.f) - log1pf(__expf(-fabsf(v)));        // log_sigmoid
  return v;                                                               // identity
}

// ---------------------------------------------------------------------------
// TDM: DMA one 2D bf16 tile (tile_rows x tile_k, row stride in elements) into LDS.
// Builds the D# per CDNA5 ISA ch.8: group0 = {count, lds_addr, global_addr, type=2},
// group1 = {data_size=2B, tensor_dim==tile_dim (no OOB), tile dims, dim0 stride}.
// Groups 2/3 zero (2D tensor). Issued by one wave; completion via TENSORcnt.
// ---------------------------------------------------------------------------
__device__ __forceinline__ void tdm_load_2d(const void* gptr, unsigned lds_off,
                                            unsigned tile_k, unsigned tile_rows,
                                            unsigned long long row_stride_elems) {
  unsigned long long ga = (unsigned long long)(size_t)gptr;
  unsigned long long s0 = row_stride_elems;          // in data_size units
  u32x4 g0;
  g0.x = 1u;                                          // count = 1 valid descriptor
  g0.y = lds_off;                                     // LDS byte address
  g0.z = (unsigned)(ga & 0xFFFFFFFFu);                // global_addr[31:0]
  g0.w = (unsigned)((ga >> 32) & 0x1FFFFFFu) | (2u << 30);  // addr[56:32] | type=2
  i32x8 g1;
  g1[0] = (int)(1u << 16);                            // data_size = 1 -> 2 bytes
  g1[1] = (int)((tile_k & 0xFFFFu) << 16);            // tensor_dim0[15:0]
  g1[2] = (int)(((tile_k >> 16) & 0xFFFFu) | ((tile_rows & 0xFFFFu) << 16)); // td0 hi | td1 lo
  g1[3] = (int)(((tile_rows >> 16) & 0xFFFFu) | (tile_k << 16));             // td1 hi | tile_dim0
  g1[4] = (int)(tile_rows & 0xFFFFu);                 // tile_dim1 (tile_dim2 = 0)
  g1[5] = (int)(unsigned)(s0 & 0xFFFFFFFFu);          // tensor_dim0_stride[31:0]
  g1[6] = (int)(unsigned)((s0 >> 32) & 0xFFFFu);      // stride[47:32] (dim1 stride unused)
  g1[7] = 0;
  i32x4 z4 = {0, 0, 0, 0};
  i32x8 z8 = {0, 0, 0, 0, 0, 0, 0, 0};
  __builtin_amdgcn_tensor_load_to_lds(g0, g1, z4, z4, z8, 0);
}

// ---------------------------------------------------------------------------
// C[m][n] = act( sum_k X[m][k] * W[n][k] )   (X: MxK bf16, W: NxK bf16, C fp32)
// grid (M/128, N/64), block 256 (8 waves). TDM stages X(128xKT) + W(64xKT) in LDS,
// waves compute bf16 WMMA 16x16x32 from LDS fragments.
// ---------------------------------------------------------------------------
template <int MODE>
__global__ __launch_bounds__(256)
void gemm_xwT_kernel(const unsigned short* __restrict__ X,
                     const unsigned short* __restrict__ W,
                     float* __restrict__ C, int M, int N, int K) {
  __shared__ unsigned short xs[128 * KT];   // 32 KB
  __shared__ unsigned short wsh[64 * KT];   // 16 KB

  const int wave = threadIdx.x >> 5;
  const int lane = threadIdx.x & 31;
  const int lm   = lane & 15;
  const int half = lane >> 4;
  const int m0   = blockIdx.x * 128 + wave * 16;
  const int n0   = blockIdx.y * 64;

  const unsigned xs_off = (unsigned)(size_t)(void*)xs;   // LDS byte offset
  const unsigned ws_off = (unsigned)(size_t)(void*)wsh;

  v8f acc[4] = {};

  for (int k0 = 0; k0 < K; k0 += KT) {
    if (threadIdx.x == 0) {
      tdm_load_2d(X + (size_t)blockIdx.x * 128 * K + k0, xs_off, KT, 128, (unsigned long long)K);
      tdm_load_2d(W + (size_t)n0 * K + k0,               ws_off, KT, 64,  (unsigned long long)K);
      __builtin_amdgcn_s_wait_tensorcnt(0);
    }
    __syncthreads();

#pragma unroll
    for (int kk = 0; kk < KT; kk += 32) {
      // A fragment (16x32): lanes 0-15 rows M, Ks {0..7,16..23}; lanes 16-31 Ks {8..15,24..31}
      const unsigned short* ar = xs + (wave * 16 + lm) * KT + kk + half * 8;
      bf16x8 alo = *(const bf16x8*)(ar);
      bf16x8 ahi = *(const bf16x8*)(ar + 16);
      bf16x16 a = __builtin_shufflevector(alo, ahi, 0,1,2,3,4,5,6,7,8,9,10,11,12,13,14,15);
#pragma unroll
      for (int j = 0; j < 4; ++j) {
        // B fragment (32x16): B[k][n] = W[n][k]; lanes 0-15 Ks {0..7,16..23}; lanes 16-31 Ks {8..15,24..31}
        const unsigned short* br = wsh + (j * 16 + lm) * KT + kk + half * 8;
        bf16x8 blo = *(const bf16x8*)(br);
        bf16x8 bhi = *(const bf16x8*)(br + 16);
        bf16x16 b = __builtin_shufflevector(blo, bhi, 0,1,2,3,4,5,6,7,8,9,10,11,12,13,14,15);
        acc[j] = __builtin_amdgcn_wmma_f32_16x16x32_bf16(false, a, false, b, (short)0, acc[j],
                                                         false, false);
      }
    }
    __syncthreads();   // all waves done with this tile before next DMA overwrites it
  }

  // D layout: vgpr i -> row m0 + i + 8*half, col n0 + j*16 + lm
#pragma unroll
  for (int j = 0; j < 4; ++j) {
#pragma unroll
    for (int i = 0; i < 8; ++i) {
      size_t row = (size_t)(m0 + i + 8 * half);
      C[row * N + n0 + j * 16 + lm] = act_fn<MODE>(acc[j][i]);
    }
  }
}

// ---------------------------------------------------------------------------
// fp32 WMMA helper (16x16x4)
// ---------------------------------------------------------------------------
__device__ __forceinline__ v8f wmma4(v2f a, v2f b, v8f c) {
  return __builtin_amdgcn_wmma_f32_16x16x4_f32(false, a, false, b, (short)0, c, false, false);
}

// ---------------------------------------------------------------------------
// Chunked GLA scan. One workgroup per (b,h). Sequential over 64 chunks.
// State 128x128 fp32 lives in LDS for the whole kernel (CDNA5: 320KB/WGP).
// ---------------------------------------------------------------------------
__global__ __launch_bounds__(256)
void gla_kernel(const float* __restrict__ Q, const float* __restrict__ LG,
                const float* __restrict__ V, float* __restrict__ O) {
  extern __shared__ float smem[];
  float* st   = smem;                 // 128*128  state[r][d]
  float* qh   = st + DF * DF;         // 64*128   q_hat[t][r]
  float* kh   = qh + CHK * DF;        // 64*128   k_hat[t][r]
  float* vv   = kh + CHK * DF;        // 64*128   v[t][d]
  float* at   = vv + CHK * DF;        // 64*64    attn[t][s]
  float* dec2 = at + CHK * CHK;       // 128      exp(total_r)

  const int b    = blockIdx.x / NH;
  const int h    = blockIdx.x % NH;
  const int col0 = h * DF;
  const int wave = threadIdx.x >> 5;
  const int lane = threadIdx.x & 31;
  const int lm   = lane & 15;
  const int half = lane >> 4;

  for (int i = threadIdx.x; i < DF * DF; i += blockDim.x) st[i] = 0.f;
  __syncthreads();

  for (int c = 0; c < NC; ++c) {
    const size_t tok0 = (size_t)b * SEQ_T + (size_t)c * CHK;

    // ---- preprocess: cumsum scan (threads 0..127) + V tile load (threads 128..255)
    if (threadIdx.x < DF) {
      const int r = threadIdx.x;
      float cum = 0.f;
      for (int t = 0; t < CHK; ++t) {
        const size_t g = (tok0 + t) * D_MODEL + col0 + r;
        float lg = LG[g];
        cum += lg;
        qh[t * DF + r] = Q[g] * __expf(cum);            // q * exp(cum)
        kh[t * DF + r] = -expm1f(lg) * __expf(-cum);    // (1-exp(lg)) * exp(-cum)
      }
      dec2[r] = __expf(cum);                            // exp(total_r)
    } else {
      const int d = threadIdx.x - DF;
      for (int t = 0; t < CHK; ++t)
        vv[t * DF + d] = V[(tok0 + t) * D_MODEL + col0 + d];
    }
    __syncthreads();

    // ---- cross = q_hat(64x128) @ state(128x128); wave w -> mt=w/2, nt=(w&1)*4+j
    const int mt  = wave >> 1;
    const int nb4 = (wave & 1) * 4;
    v8f oacc[4] = {};
    {
      const int arow = mt * 16 + lm;
      for (int k0 = 0; k0 < DF; k0 += 4) {
        v2f a = *(const v2f*)&qh[arow * DF + k0 + 2 * half];
#pragma unroll
        for (int j = 0; j < 4; ++j) {
          const int n0 = (nb4 + j) * 16;
          v2f bb;
          bb.x = st[(k0 + 2 * half) * DF + n0 + lm];
          bb.y = st[(k0 + 2 * half + 1) * DF + n0 + lm];
          oacc[j] = wmma4(a, bb, oacc[j]);
        }
      }
    }

    // ---- attn = q_hat @ k_hat^T (64x64), causal masked; wave gets 2 of 16 tiles
#pragma unroll
    for (int j2 = 0; j2 < 2; ++j2) {
      const int ta = wave * 2 + j2;
      const int amt = ta >> 2, ant = ta & 3;
      v8f aacc = {};
      const int arow = amt * 16 + lm;
      const int brow = ant * 16 + lm;
      for (int k0 = 0; k0 < DF; k0 += 4) {
        v2f a  = *(const v2f*)&qh[arow * DF + k0 + 2 * half];
        v2f bb = *(const v2f*)&kh[brow * DF + k0 + 2 * half];
        aacc = wmma4(a, bb, aacc);
      }
#pragma unroll
      for (int i = 0; i < 8; ++i) {
        int trow = amt * 16 + i + 8 * half;
        int scol = ant * 16 + lm;
        at[trow * CHK + scol] = (scol <= trow) ? aacc[i] : 0.f;
      }
    }
    __syncthreads();

    // ---- intra = attn @ v, add to cross, write O
    {
      const int arow = mt * 16 + lm;
      for (int k0 = 0; k0 < CHK; k0 += 4) {
        v2f a = *(const v2f*)&at[arow * CHK + k0 + 2 * half];
#pragma unroll
        for (int j = 0; j < 4; ++j) {
          const int n0 = (nb4 + j) * 16;
          v2f bb;
          bb.x = vv[(k0 + 2 * half) * DF + n0 + lm];
          bb.y = vv[(k0 + 2 * half + 1) * DF + n0 + lm];
          oacc[j] = wmma4(a, bb, oacc[j]);
        }
      }
#pragma unroll
      for (int j = 0; j < 4; ++j) {
        const int n0 = (nb4 + j) * 16;
#pragma unroll
        for (int i = 0; i < 8; ++i) {
          int trow = mt * 16 + i + 8 * half;
          O[(tok0 + trow) * D_MODEL + col0 + n0 + lm] = oacc[j][i];
        }
      }
    }
    __syncthreads();

    // ---- state update: state = state*exp(total_r) + kg^T @ v,
    //      kg[t][r] = kh[t][r]*exp(total_r). Wave owns row-tile smt=wave, all 8 col-tiles.
    {
      v8f sacc[8] = {};
      const int r = wave * 16 + lm;
      const float d2 = dec2[r];
      for (int k0 = 0; k0 < CHK; k0 += 4) {
        v2f a;
        a.x = kh[(k0 + 2 * half) * DF + r] * d2;
        a.y = kh[(k0 + 2 * half + 1) * DF + r] * d2;
#pragma unroll
        for (int j = 0; j < 8; ++j) {
          const int n0 = j * 16;
          v2f bb;
          bb.x = vv[(k0 + 2 * half) * DF + n0 + lm];
          bb.y = vv[(k0 + 2 * half + 1) * DF + n0 + lm];
          sacc[j] = wmma4(a, bb, sacc[j]);
        }
      }
#pragma unroll
      for (int j = 0; j < 8; ++j) {
#pragma unroll
        for (int i = 0; i < 8; ++i) {
          int rr = wave * 16 + i + 8 * half;
          int idx = rr * DF + j * 16 + lm;
          st[idx] = st[idx] * dec2[rr] + sacc[j][i];
        }
      }
    }
    __syncthreads();
  }
}

// ---------------------------------------------------------------------------
// RMS norm over D=1024 per token, write bf16 for the final GEMM
// ---------------------------------------------------------------------------
__global__ __launch_bounds__(256)
void rmsnorm_kernel(const float* __restrict__ O, const float* __restrict__ gw,
                    unsigned short* __restrict__ Ob) {
  __shared__ float red[256];
  const size_t tok = blockIdx.x;
  float vals[4];
  float s = 0.f;
#pragma unroll
  for (int i = 0; i < 4; ++i) {
    vals[i] = O[tok * D_MODEL + threadIdx.x + i * 256];
    s += vals[i] * vals[i];
  }
  red[threadIdx.x] = s;
  __syncthreads();
  for (int off = 128; off > 0; off >>= 1) {
    if (threadIdx.x < (unsigned)off) red[threadIdx.x] += red[threadIdx.x + off];
    __syncthreads();
  }
  const float norm = rsqrtf(red[0] / (float)D_MODEL + 1e-5f);
#pragma unroll
  for (int i = 0; i < 4; ++i) {
    int d = threadIdx.x + i * 256;
    Ob[tok * D_MODEL + d] = f2bf(vals[i] * norm * gw[d]);
  }
}

// ---------------------------------------------------------------------------
// Launch
// ---------------------------------------------------------------------------
extern "C" void kernel_launch(void* const* d_in, const int* in_sizes, int n_in,
                              void* d_out, int out_size, void* d_ws, size_t ws_size,
                              hipStream_t stream) {
  (void)in_sizes; (void)n_in; (void)out_size; (void)ws_size;
  const float* x  = (const float*)d_in[0];
  const float* Wq = (const float*)d_in[1];
  const float* Wf = (const float*)d_in[2];
  const float* Wi = (const float*)d_in[3];
  const float* Wo = (const float*)d_in[4];
  const float* gw = (const float*)d_in[5];

  const size_t MT = (size_t)NB * SEQ_T;           // 16384 tokens
  const size_t DD = (size_t)D_MODEL * D_MODEL;

  float* Q  = (float*)d_ws;                       // MT*D fp32
  float* LG = Q  + MT * D_MODEL;
  float* V  = LG + MT * D_MODEL;
  float* O  = V  + MT * D_MODEL;
  unsigned short* xb  = (unsigned short*)(O + MT * D_MODEL);
  unsigned short* Wqb = xb  + MT * D_MODEL;
  unsigned short* Wfb = Wqb + DD;
  unsigned short* Wib = Wfb + DD;
  unsigned short* Wob = Wib + DD;
  unsigned short* Ob  = Wob + DD;                 // total ~328 MB

  cvt_bf16_kernel<<<2048, 256, 0, stream>>>(x,  xb,  (int)(MT * D_MODEL));
  cvt_bf16_kernel<<<512,  256, 0, stream>>>(Wq, Wqb, (int)DD);
  cvt_bf16_kernel<<<512,  256, 0, stream>>>(Wf, Wfb, (int)DD);
  cvt_bf16_kernel<<<512,  256, 0, stream>>>(Wi, Wib, (int)DD);
  cvt_bf16_kernel<<<512,  256, 0, stream>>>(Wo, Wob, (int)DD);

  dim3 gg((unsigned)(MT / 128), (unsigned)(D_MODEL / 64));
  gemm_xwT_kernel<0><<<gg, 256, 0, stream>>>(xb, Wqb, Q,  (int)MT, D_MODEL, D_MODEL);
  gemm_xwT_kernel<1><<<gg, 256, 0, stream>>>(xb, Wfb, LG, (int)MT, D_MODEL, D_MODEL);
  gemm_xwT_kernel<2><<<gg, 256, 0, stream>>>(xb, Wib, V,  (int)MT, D_MODEL, D_MODEL);

  const int gla_smem = (DF * DF + 3 * CHK * DF + CHK * CHK + DF) * (int)sizeof(float); // 180736 B
  (void)hipFuncSetAttribute((const void*)gla_kernel,
                            hipFuncAttributeMaxDynamicSharedMemorySize, gla_smem);
  gla_kernel<<<NB * NH, 256, gla_smem, stream>>>(Q, LG, V, O);

  rmsnorm_kernel<<<(unsigned)MT, 256, 0, stream>>>(O, gw, Ob);

  gemm_xwT_kernel<2><<<gg, 256, 0, stream>>>(Ob, Wob, (float*)d_out, (int)MT, D_MODEL, D_MODEL);
}
